// LSTMModel_15874199126102
// MI455X (gfx1250) — compile-verified, matrix operand order checked by reading, not run on hardware
//
#include <hip/hip_runtime.h>
#include <hip/hip_bf16.h>

// ---------------------------------------------------------------------------
// 2-layer LSTM (B=256, T=512, I=64, H=512) + FC(512->1) for MI455X (gfx1250).
// bf16 V_WMMA_F32_16X16X32_BF16 for every GEMM, f32 accumulation.
// Both layers advanced in one time loop (no [B,T,H] sequence buffer).
// Each wave owns a 32(M) x 16(H) tile x 4 gates: every B fragment is reused
// by 2 WMMAs, cutting vector-load pressure to 1.5 loads/WMMA, and the fused
// cell update runs in-register right after the K loop.
// ---------------------------------------------------------------------------

typedef __bf16 bf16x8  __attribute__((ext_vector_type(8)));
typedef __bf16 bf16x16 __attribute__((ext_vector_type(16)));
typedef float  f32x8   __attribute__((ext_vector_type(8)));

constexpr int B_ = 256;
constexpr int T_ = 512;
constexpr int I_ = 64;
constexpr int H_ = 512;
constexpr int G_ = 4 * H_;   // 2048 gate outputs

__device__ __forceinline__ unsigned short f2bf(float f) {
    unsigned int u = __builtin_bit_cast(unsigned int, f);
    unsigned int r = u + 0x7FFFu + ((u >> 16) & 1u);   // round-to-nearest-even
    return (unsigned short)(r >> 16);
}
__device__ __forceinline__ float bf2f(unsigned short s) {
    unsigned int u = ((unsigned int)s) << 16;
    return __builtin_bit_cast(float, u);
}
__device__ __forceinline__ float sigmoid_(float x) {
    return 1.0f / (1.0f + __expf(-x));
}

// Load a 16x32 bf16 A/B fragment row-segment for this lane.
// ISA 7.12.2: lanes 0-15 / 16-31 split K as [khalf*8 .. +8) and [16+khalf*8 .. +8).
__device__ __forceinline__ bf16x16 load_frag(const unsigned short* p) {
    bf16x8 lo = *reinterpret_cast<const bf16x8*>(p);
    bf16x8 hi = *reinterpret_cast<const bf16x8*>(p + 16);
    return __builtin_shufflevector(lo, hi, 0, 1, 2, 3, 4, 5, 6, 7,
                                           8, 9, 10, 11, 12, 13, 14, 15);
}

// ---------------------------------------------------------------------------
// One LSTM timestep for one layer:
//   gates[b, g*512+n] = hPrev[b,:]@Whh[g*512+n,:] + xIn[b,:]@Wih[g*512+n,:] + bias
//   c = sigm(f)*c + sigm(i)*tanh(g);  hNew = sigm(o)*tanh(c)
// Wave tile: 32 batch rows (2 M-tiles) x 16 hidden cols x 4 gates.
// Waves in a block share tileH -> identical B fragments hit in WGP$.
// Grid: 128 blocks x 64 threads (2 waves); 8 M-groups x 32 H-tiles.
// ---------------------------------------------------------------------------
__global__ __launch_bounds__(64) void lstm_step_kernel(
    const unsigned short* __restrict__ hPrev,  // [256,512] bf16
    const unsigned short* __restrict__ xIn,    // [256,ldX] bf16 (first kIn cols)
    int ldX, int kIn,
    const unsigned short* __restrict__ Whh,    // [2048,512] bf16 row-major
    const unsigned short* __restrict__ Wih,    // [2048,kIn] bf16 row-major
    const float* __restrict__ bias,            // [2048] = bih+bhh
    float* __restrict__ c,                     // [256,512] f32 (in-place)
    unsigned short* __restrict__ hNew)         // [256,512] bf16
{
    const int wave  = threadIdx.x >> 5;
    const int lane  = threadIdx.x & 31;
    const int tileH = blockIdx.x & 31;                   // 0..31 hidden-col tile
    const int mg    = ((blockIdx.x >> 5) << 1) | wave;   // 0..7 M-group (32 rows)
    const int col   = lane & 15;
    const int khalf = lane >> 4;

    const int rowA0 = mg * 32 + col;        // batch row, first 16-row half
    const int rowA1 = rowA0 + 16;           // batch row, second 16-row half
    const int hcol  = tileH * 16 + col;     // hidden column (B row select, C col)

    f32x8 acc[2][4];
#pragma unroll
    for (int g = 0; g < 4; ++g) {
        float bv = bias[g * H_ + hcol];
        acc[0][g] = (f32x8){bv, bv, bv, bv, bv, bv, bv, bv};
        acc[1][g] = (f32x8){bv, bv, bv, bv, bv, bv, bv, bv};
    }

    // ---- recurrent part: K over hidden state (512) ----
    const unsigned short* aBase0 = hPrev + (size_t)rowA0 * H_;
    const unsigned short* aBase1 = hPrev + (size_t)rowA1 * H_;
    for (int k = 0; k < H_; k += 32) {
        const int ko = k + khalf * 8;
        __builtin_prefetch(aBase0 + ko + 64, 0, 1);      // global_prefetch_b8
        __builtin_prefetch(aBase1 + ko + 64, 0, 1);
        bf16x16 a0 = load_frag(aBase0 + ko);
        bf16x16 a1 = load_frag(aBase1 + ko);
#pragma unroll
        for (int g = 0; g < 4; ++g) {
            const unsigned short* bp =
                Whh + (size_t)(g * H_ + hcol) * H_ + ko;
            bf16x16 b = load_frag(bp);
            acc[0][g] = __builtin_amdgcn_wmma_f32_16x16x32_bf16(
                false, a0, false, b, (short)0, acc[0][g], false, false);
            acc[1][g] = __builtin_amdgcn_wmma_f32_16x16x32_bf16(
                false, a1, false, b, (short)0, acc[1][g], false, false);
        }
    }

    // ---- input part: K over layer input (64 or 512) ----
    const unsigned short* xBase0 = xIn + (size_t)rowA0 * ldX;
    const unsigned short* xBase1 = xIn + (size_t)rowA1 * ldX;
    for (int k = 0; k < kIn; k += 32) {
        const int ko = k + khalf * 8;
        bf16x16 a0 = load_frag(xBase0 + ko);
        bf16x16 a1 = load_frag(xBase1 + ko);
#pragma unroll
        for (int g = 0; g < 4; ++g) {
            const unsigned short* bp =
                Wih + (size_t)(g * H_ + hcol) * kIn + ko;
            bf16x16 b = load_frag(bp);
            acc[0][g] = __builtin_amdgcn_wmma_f32_16x16x32_bf16(
                false, a0, false, b, (short)0, acc[0][g], false, false);
            acc[1][g] = __builtin_amdgcn_wmma_f32_16x16x32_bf16(
                false, a1, false, b, (short)0, acc[1][g], false, false);
        }
    }

    // ---- fused cell update (C/D layout: VGPR r -> M = r + 8*khalf) ----
#pragma unroll
    for (int half = 0; half < 2; ++half) {
        const int mbase = mg * 32 + half * 16;
#pragma unroll
        for (int r = 0; r < 8; ++r) {
            const int b = mbase + r + khalf * 8;          // global batch row
            const size_t ix = (size_t)b * H_ + hcol;
            float iv = sigmoid_(acc[half][0][r]);
            float fv = sigmoid_(acc[half][1][r]);
            float gv = tanhf(acc[half][2][r]);
            float ov = sigmoid_(acc[half][3][r]);
            float cn = fv * c[ix] + iv * gv;
            c[ix] = cn;
            hNew[ix] = f2bf(ov * tanhf(cn));
        }
    }
}

// ---------------------------------------------------------------------------
// Helpers: f32 -> bf16 convert, bias sum, zero-fill, final FC.
// ---------------------------------------------------------------------------
__global__ void f32_to_bf16_kernel(const float* __restrict__ in,
                                   unsigned short* __restrict__ out, int n) {
    int i = blockIdx.x * blockDim.x + threadIdx.x;
    int s = gridDim.x * blockDim.x;
    for (; i < n; i += s) out[i] = f2bf(in[i]);
}

__global__ void add_bias_kernel(const float* __restrict__ a,
                                const float* __restrict__ b,
                                float* __restrict__ o, int n) {
    int i = blockIdx.x * blockDim.x + threadIdx.x;
    if (i < n) o[i] = a[i] + b[i];
}

__global__ void zero_u32_kernel(unsigned int* __restrict__ p, int n) {
    int i = blockIdx.x * blockDim.x + threadIdx.x;
    int s = gridDim.x * blockDim.x;
    for (; i < n; i += s) p[i] = 0u;
}

__global__ void fc_kernel(const unsigned short* __restrict__ h2,  // [256,512] bf16
                          const float* __restrict__ Wfc,          // [512]
                          const float* __restrict__ bfc,          // [1]
                          float* __restrict__ out) {              // [256]
    int b = threadIdx.x;   // 256 threads
    float s = bfc[0];
    for (int h = 0; h < H_; ++h) s += bf2f(h2[(size_t)b * H_ + h]) * Wfc[h];
    out[b] = s;
}

// ---------------------------------------------------------------------------
// Workspace layout (bytes, all 256B-aligned offsets), total ~25.4 MB
// ---------------------------------------------------------------------------
extern "C" void kernel_launch(void* const* d_in, const int* in_sizes, int n_in,
                              void* d_out, int out_size, void* d_ws, size_t ws_size,
                              hipStream_t stream) {
    const float* x    = (const float*)d_in[0];
    const float* Wih0 = (const float*)d_in[1];
    const float* Whh0 = (const float*)d_in[2];
    const float* bih0 = (const float*)d_in[3];
    const float* bhh0 = (const float*)d_in[4];
    const float* Wih1 = (const float*)d_in[5];
    const float* Whh1 = (const float*)d_in[6];
    const float* bih1 = (const float*)d_in[7];
    const float* bhh1 = (const float*)d_in[8];
    const float* Wfc  = (const float*)d_in[9];
    const float* bfc  = (const float*)d_in[10];
    float* out = (float*)d_out;

    char* ws = (char*)d_ws;
    size_t off = 0;
    auto take = [&](size_t bytes) { char* p = ws + off; off += bytes; return p; };

    unsigned short* xbf   = (unsigned short*)take((size_t)B_ * T_ * I_ * 2); // 16 MB
    unsigned short* whh0  = (unsigned short*)take((size_t)G_ * H_ * 2);      // 2 MB
    unsigned short* wih0  = (unsigned short*)take((size_t)G_ * I_ * 2);      // 256 KB
    unsigned short* whh1  = (unsigned short*)take((size_t)G_ * H_ * 2);      // 2 MB
    unsigned short* wih1  = (unsigned short*)take((size_t)G_ * H_ * 2);      // 2 MB
    float*          bias0 = (float*)take((size_t)G_ * 4);
    float*          bias1 = (float*)take((size_t)G_ * 4);
    unsigned short* h1b0  = (unsigned short*)take((size_t)B_ * H_ * 2);
    unsigned short* h1b1  = (unsigned short*)take((size_t)B_ * H_ * 2);
    unsigned short* h2b0  = (unsigned short*)take((size_t)B_ * H_ * 2);
    unsigned short* h2b1  = (unsigned short*)take((size_t)B_ * H_ * 2);
    float*          c1    = (float*)take((size_t)B_ * H_ * 4);
    float*          c2    = (float*)take((size_t)B_ * H_ * 4);
    if (ws_size < off) return;   // workspace too small; nothing safe to do

    // --- precompute: bf16 conversions + combined biases + state zeroing ---
    f32_to_bf16_kernel<<<8192, 256, 0, stream>>>(x, xbf, B_ * T_ * I_);
    f32_to_bf16_kernel<<<2048, 256, 0, stream>>>(Whh0, whh0, G_ * H_);
    f32_to_bf16_kernel<<<512,  256, 0, stream>>>(Wih0, wih0, G_ * I_);
    f32_to_bf16_kernel<<<2048, 256, 0, stream>>>(Whh1, whh1, G_ * H_);
    f32_to_bf16_kernel<<<2048, 256, 0, stream>>>(Wih1, wih1, G_ * H_);
    add_bias_kernel<<<8, 256, 0, stream>>>(bih0, bhh0, bias0, G_);
    add_bias_kernel<<<8, 256, 0, stream>>>(bih1, bhh1, bias1, G_);
    zero_u32_kernel<<<64,  256, 0, stream>>>((unsigned int*)h1b0, B_ * H_ / 2);
    zero_u32_kernel<<<64,  256, 0, stream>>>((unsigned int*)h2b0, B_ * H_ / 2);
    zero_u32_kernel<<<128, 256, 0, stream>>>((unsigned int*)c1, B_ * H_);
    zero_u32_kernel<<<128, 256, 0, stream>>>((unsigned int*)c2, B_ * H_);

    // --- both layers advanced in the same time loop (no [B,T,H] buffer) ---
    for (int t = 0; t < T_; ++t) {
        const unsigned short* h1p = (t & 1) ? h1b1 : h1b0;
        unsigned short*       h1n = (t & 1) ? h1b0 : h1b1;
        const unsigned short* h2p = (t & 1) ? h2b1 : h2b0;
        unsigned short*       h2n = (t & 1) ? h2b0 : h2b1;
        // layer 0: input is x[:, t, :]  (row stride T*I, K=64)
        lstm_step_kernel<<<128, 64, 0, stream>>>(
            h1p, xbf + (size_t)t * I_, T_ * I_, I_, whh0, wih0, bias0, c1, h1n);
        // layer 1: input is h1[t] just produced (K=512)
        lstm_step_kernel<<<128, 64, 0, stream>>>(
            h2p, h1n, H_, H_, whh1, wih1, bias1, c2, h2n);
    }

    // final h2 written at t=511 (odd) -> h2b0
    fc_kernel<<<1, 256, 0, stream>>>(h2b0, Wfc, bfc, out);
}
